// LSTM_Attention_41111426957829
// MI455X (gfx1250) — matrix operand order repaired and checked by reading, not compiled, not run on hardware
//
#include <hip/hip_runtime.h>
#include <hip/hip_bf16.h>

// ---------------------------------------------------------------------------
// Problem constants (from reference)
// ---------------------------------------------------------------------------
static constexpr int B  = 128;
static constexpr int L  = 196;
static constexpr int F2 = 512;
static constexpr int T  = 32;
static constexpr int D  = 256;
static constexpr int H  = 512;
static constexpr int KD = D + F2 + H;   // 1280 : combined GEMM K (word|context|h)
static constexpr int N4H = 4 * H;       // 2048

typedef __bf16 bf16;
typedef __attribute__((ext_vector_type(16))) __bf16 v16bf;
typedef __attribute__((ext_vector_type(8)))  float  v8f;

union Frag16 { uint4 u[2]; v16bf v; };

// ---------------------------------------------------------------------------
// bf16 WMMA GEMM, no LDS, software-pipelined: C = A * Bt^T (+bias), fp32 acc.
//   A  : row-major [M, K]      (lda = row stride in elements)
//   Bt : column-major [N, K]   (each output column's K values contiguous)
// All operands are L2-resident; fragments load straight from global.
//   A frag (16x32): lane(h,ln) -> k = {8h..8h+7, 16+8h..16+8h+7}, row = ln
//   B frag (32x16): lane(h,ln) -> k = {8h..8h+7, 16+8h..16+8h+7}, col = ln
// Block = 8 waves; wave tile 16(M) x 64(N); block tile 64 x 128.
// Next K-slice's 10 loads are issued before the current slice's 4 WMMAs so
// load latency overlaps WMMA execution (partial s_wait_loadcnt, not drains).
// Requires M%64==0, N%128==0, K%32==0. EXEC stays all-ones (no divergence).
// ---------------------------------------------------------------------------
__global__ __launch_bounds__(256)
void wmma_gemm_bf16(const bf16* __restrict__ A, int lda,
                    const bf16* __restrict__ Bt,
                    const float* __restrict__ bias,
                    float* __restrict__ C, int ldc, int K)
{
    const int lane = threadIdx.x & 31;
    const int wave = threadIdx.x >> 5;
    const int m_blk = blockIdx.x * 64;
    const int n_blk = blockIdx.y * 128;

    const int m0 = (wave >> 1) * 16;
    const int n0 = (wave & 1) * 64;
    const int h  = lane >> 4;
    const int ln = lane & 15;

    const bf16* arow = A  + (size_t)(m_blk + m0 + ln) * lda + 8 * h;
    const bf16* bcol = Bt + (size_t)(n_blk + n0 + ln) * K + 8 * h;

    v8f acc[4] = {};

    // prologue: fetch K-slice 0
    Frag16 fa, fb[4];
    fa.u[0] = *(const uint4*)(arow);
    fa.u[1] = *(const uint4*)(arow + 16);
#pragma unroll
    for (int s = 0; s < 4; ++s) {
        const bf16* bp = bcol + (size_t)s * 16 * K;
        fb[s].u[0] = *(const uint4*)(bp);
        fb[s].u[1] = *(const uint4*)(bp + 16);
    }

#pragma unroll 2
    for (int k0 = 32; k0 < K; k0 += 32) {
        // prefetch next K-slice while current WMMAs execute
        Frag16 na, nb[4];
        na.u[0] = *(const uint4*)(arow + k0);
        na.u[1] = *(const uint4*)(arow + k0 + 16);
#pragma unroll
        for (int s = 0; s < 4; ++s) {
            const bf16* bp = bcol + (size_t)s * 16 * K + k0;
            nb[s].u[0] = *(const uint4*)(bp);
            nb[s].u[1] = *(const uint4*)(bp + 16);
        }
#pragma unroll
        for (int s = 0; s < 4; ++s)
            acc[s] = __builtin_amdgcn_wmma_f32_16x16x32_bf16(
                false, fa.v, false, fb[s].v, (short)0, acc[s], false, false);
        fa = na;
#pragma unroll
        for (int s = 0; s < 4; ++s) fb[s] = nb[s];
    }
    // drain last slice
#pragma unroll
    for (int s = 0; s < 4; ++s)
        acc[s] = __builtin_amdgcn_wmma_f32_16x16x32_bf16(
            false, fa.v, false, fb[s].v, (short)0, acc[s], false, false);

    // epilogue: C layout = 8 VGPRs, row = m0 + 8h + r, col = lane&15
#pragma unroll
    for (int s = 0; s < 4; ++s) {
        int col = n_blk + n0 + s * 16 + ln;
        float bv = bias ? bias[col] : 0.0f;
#pragma unroll
        for (int r = 0; r < 8; ++r) {
            int row = m_blk + m0 + h * 8 + r;
            C[(size_t)row * ldc + col] = acc[s][r] + bv;
        }
    }
}

// ---------------------------------------------------------------------------
// Setup kernels
// ---------------------------------------------------------------------------
// Combined LSTM weight, bf16, COLUMN-major: WcT[col * 1280 + row]
__global__ __launch_bounds__(256)
void build_wc(const float* __restrict__ lstm_kernel,
              const float* __restrict__ lstm_recurrent,
              bf16* __restrict__ WcT)
{
    size_t i = (size_t)blockIdx.x * 256 + threadIdx.x;   // over 2048*1280
    int col = (int)(i / KD);
    int row = (int)(i % KD);
    float v = (row < D + F2) ? lstm_kernel[(size_t)row * N4H + col]
                             : lstm_recurrent[(size_t)(row - (D + F2)) * N4H + col];
    WcT[i] = (bf16)v;
}

// Init-state weight [w_h | w_c], bf16, COLUMN-major: WhcT[col * 512 + k]
__global__ __launch_bounds__(256)
void build_whc(const float* __restrict__ w_h, const float* __restrict__ w_c,
               bf16* __restrict__ WhcT)
{
    size_t i = (size_t)blockIdx.x * 256 + threadIdx.x;   // over 1024*512
    int col = (int)(i >> 9);
    int k   = (int)(i & 511);
    float v = (col < H) ? w_h[(size_t)k * H + col] : w_c[(size_t)k * H + (col - H)];
    WhcT[i] = (bf16)v;
}

__global__ __launch_bounds__(256)
void convert_input(const float* __restrict__ input_data, bf16* __restrict__ input_bf)
{
    size_t i = (size_t)blockIdx.x * 256 + threadIdx.x;   // over B*L*F2
    input_bf[i] = (bf16)input_data[i];
}

__global__ __launch_bounds__(256)
void mean_kernel(const float* __restrict__ input_data, bf16* __restrict__ Amean)
{
    int b = blockIdx.x;
    int f = blockIdx.y * 256 + threadIdx.x;
    const float* ip = input_data + (size_t)b * L * F2 + f;
    float acc = 0.f;
    for (int l = 0; l < L; ++l) acc += ip[(size_t)l * F2];
    Amean[(size_t)b * F2 + f] = (bf16)(acc * (1.0f / (float)L));
}

__global__ __launch_bounds__(256)
void featscore_kernel(const float* __restrict__ input_data, const float* __restrict__ w_a,
                      const float* __restrict__ b_a, float* __restrict__ feat_score)
{
    int gw   = blockIdx.x * 8 + (threadIdx.x >> 5);   // flat (b,l), 25088 waves
    int lane = threadIdx.x & 31;
    const float* ip = input_data + (size_t)gw * F2;
    float acc = 0.f;
    for (int k = lane; k < F2; k += 32) acc = fmaf(ip[k], w_a[k], acc);
    for (int off = 16; off > 0; off >>= 1) acc += __shfl_xor(acc, off, 32);
    if (lane == 0) feat_score[gw] = acc + b_a[0];
}

__global__ __launch_bounds__(256)
void init_state(const float* __restrict__ hc_raw,   // [B, 1024]
                const float* __restrict__ b_h, const float* __restrict__ b_c,
                float* __restrict__ h_state, float* __restrict__ c_state,
                bf16* __restrict__ Abuf)
{
    int i = blockIdx.x * 256 + threadIdx.x;   // 0..B*H
    int b = i >> 9, j = i & 511;
    float hv = tanhf(hc_raw[(size_t)b * 1024 + j] + b_h[j]);
    float cv = tanhf(hc_raw[(size_t)b * 1024 + H + j] + b_c[j]);
    h_state[i] = hv;
    c_state[i] = cv;
    Abuf[(size_t)b * KD + D + F2 + j] = (bf16)hv;
}

// ---------------------------------------------------------------------------
// Per-step kernels
// ---------------------------------------------------------------------------
__global__ __launch_bounds__(256)
void attn_softmax(const float* __restrict__ h_state, const float* __restrict__ w_a_h,
                  const float* __restrict__ feat_score,
                  float* __restrict__ alfa, float* __restrict__ alfas_out, int t)
{
    __shared__ float red[256];
    __shared__ float bcast;
    const int b = blockIdx.x, tid = threadIdx.x;

    float p = h_state[b * H + tid] * w_a_h[tid]
            + h_state[b * H + 256 + tid] * w_a_h[256 + tid];
    red[tid] = p; __syncthreads();
    for (int off = 128; off > 0; off >>= 1) { if (tid < off) red[tid] += red[tid + off]; __syncthreads(); }
    if (tid == 0) bcast = red[0];
    __syncthreads();
    float s = bcast;

    float sc = -INFINITY;
    if (tid < L) sc = tanhf(s + feat_score[b * L + tid]);
    red[tid] = sc; __syncthreads();
    for (int off = 128; off > 0; off >>= 1) { if (tid < off) red[tid] = fmaxf(red[tid], red[tid + off]); __syncthreads(); }
    if (tid == 0) bcast = red[0];
    __syncthreads();
    float mx = bcast;
    __syncthreads();

    float e = (tid < L) ? __expf(sc - mx) : 0.f;
    red[tid] = e; __syncthreads();
    for (int off = 128; off > 0; off >>= 1) { if (tid < off) red[tid] += red[tid + off]; __syncthreads(); }
    if (tid == 0) bcast = red[0];
    __syncthreads();
    float inv = 1.0f / bcast;

    if (tid < L) {
        float a = e * inv;
        alfa[b * L + tid] = a;
        alfas_out[(size_t)b * T * L + (size_t)t * L + tid] = a;
    }
}

// fp32-input variant (fallback when ws is too small for the bf16 input copy)
__global__ __launch_bounds__(256)
void context_kernel_f32(const float* __restrict__ input_data, const float* __restrict__ alfa,
                        bf16* __restrict__ Abuf)
{
    const int b = blockIdx.x;
    const int f = blockIdx.y * 256 + threadIdx.x;
    const float* ip = input_data + (size_t)b * L * F2 + f;
    const float* al = alfa + b * L;
    float acc = 0.f;
#pragma unroll 4
    for (int l = 0; l < L; ++l) acc = fmaf(al[l], ip[(size_t)l * F2], acc);
    Abuf[(size_t)b * KD + D + f] = (bf16)acc;   // cols 256..767
}

// bf16-input variant: halves the per-step L2 traffic (25.7 MB vs 51.4 MB)
__global__ __launch_bounds__(256)
void context_kernel_bf(const bf16* __restrict__ input_bf, const float* __restrict__ alfa,
                       bf16* __restrict__ Abuf)
{
    const int b = blockIdx.x;
    const int f = 2 * threadIdx.x;           // 512 features via 256 threads
    const bf16* ip = input_bf + (size_t)b * L * F2 + f;
    const float* al = alfa + b * L;
    float a0 = 0.f, a1 = 0.f;
#pragma unroll 4
    for (int l = 0; l < L; ++l) {
        float w = al[l];
        a0 = fmaf(w, (float)ip[(size_t)l * F2 + 0], a0);
        a1 = fmaf(w, (float)ip[(size_t)l * F2 + 1], a1);
    }
    Abuf[(size_t)b * KD + D + f + 0] = (bf16)a0;
    Abuf[(size_t)b * KD + D + f + 1] = (bf16)a1;
}

__global__ __launch_bounds__(256)
void word_convert(const float* __restrict__ word_inputs, bf16* __restrict__ Abuf, int t)
{
    int i = blockIdx.x * 256 + threadIdx.x;   // 0..B*D
    int b = i >> 8, d = i & 255;
    Abuf[(size_t)b * KD + d] = (bf16)word_inputs[(size_t)b * T * D + (size_t)t * D + d];
}

__global__ __launch_bounds__(256)
void gate_kernel(const float* __restrict__ z, float* __restrict__ c_state,
                 float* __restrict__ h_state, bf16* __restrict__ Abuf,
                 float* __restrict__ hiddens, float* __restrict__ memory, int t)
{
    int i = blockIdx.x * 256 + threadIdx.x;   // 0..B*H
    int b = i >> 9, j = i & 511;
    const float* zb = z + (size_t)b * N4H;
    float zi = zb[j], zf = zb[H + j], zg = zb[2 * H + j], zo = zb[3 * H + j];
    float si = 1.f / (1.f + __expf(-zi));
    float sf = 1.f / (1.f + __expf(-zf));
    float so = 1.f / (1.f + __expf(-zo));
    float cn = sf * c_state[i] + si * tanhf(zg);
    float hn = so * tanhf(cn);
    c_state[i] = cn;
    h_state[i] = hn;
    Abuf[(size_t)b * KD + D + F2 + j] = (bf16)hn;
    hiddens[(size_t)b * T * H + (size_t)t * H + j] = hn;
    memory[(size_t)t * B * H + (size_t)b * H + j] = cn;
}

// ---------------------------------------------------------------------------
// Orchestration
// ---------------------------------------------------------------------------
extern "C" void kernel_launch(void* const* d_in, const int* in_sizes, int n_in,
                              void* d_out, int out_size, void* d_ws, size_t ws_size,
                              hipStream_t stream)
{
    const float* input_data     = (const float*)d_in[0];
    const float* word_inputs    = (const float*)d_in[1];
    const float* w_h            = (const float*)d_in[2];
    const float* b_h            = (const float*)d_in[3];
    const float* w_c            = (const float*)d_in[4];
    const float* b_c            = (const float*)d_in[5];
    const float* w_a_h          = (const float*)d_in[6];
    const float* w_a            = (const float*)d_in[7];
    const float* b_a            = (const float*)d_in[8];
    const float* lstm_kernel    = (const float*)d_in[9];
    const float* lstm_recurrent = (const float*)d_in[10];
    const float* lstm_bias      = (const float*)d_in[11];

    float* out     = (float*)d_out;
    float* hiddens = out;                                   // [B,T,H]
    float* memory  = out + (size_t)B * T * H;               // [T,B,H]
    float* alfas   = out + (size_t)2 * B * T * H;           // [B,T,L]

    // workspace layout (all region sizes multiples of 256 B)
    char* ws = (char*)d_ws;
    size_t off = 0;
    bf16*  WcT     = (bf16*)(ws + off); off += (size_t)N4H * KD * 2;      // 5.24 MB
    bf16*  WhcT    = (bf16*)(ws + off); off += (size_t)1024 * F2 * 2;     // 1.05 MB
    bf16*  Amean   = (bf16*)(ws + off); off += (size_t)B * F2 * 2;
    bf16*  Abuf    = (bf16*)(ws + off); off += (size_t)B * KD * 2;
    float* featsc  = (float*)(ws + off); off += (size_t)B * L * 4;
    float* alfa    = (float*)(ws + off); off += (size_t)B * L * 4;
    float* h_state = (float*)(ws + off); off += (size_t)B * H * 4;
    float* c_state = (float*)(ws + off); off += (size_t)B * H * 4;
    float* hc_raw  = (float*)(ws + off); off += (size_t)B * 1024 * 4;
    float* zbuf    = (float*)(ws + off); off += (size_t)B * N4H * 4;
    bf16*  input_bf = (bf16*)(ws + off);
    size_t need_bf = off + (size_t)B * L * F2 * 2;          // +25.7 MB
    const bool use_bf_input = (ws_size >= need_bf);
    (void)in_sizes; (void)n_in; (void)out_size;

    // ---- setup ----
    build_wc <<<(KD * N4H) / 256, 256, 0, stream>>>(lstm_kernel, lstm_recurrent, WcT);
    build_whc<<<(F2 * 1024) / 256, 256, 0, stream>>>(w_h, w_c, WhcT);
    mean_kernel<<<dim3(B, F2 / 256), 256, 0, stream>>>(input_data, Amean);
    featscore_kernel<<<(B * L) / 8, 256, 0, stream>>>(input_data, w_a, b_a, featsc);
    if (use_bf_input)
        convert_input<<<(B * L * F2) / 256, 256, 0, stream>>>(input_data, input_bf);

    // h0/c0 GEMM: [128,512] x [512,1024] -> hc_raw
    wmma_gemm_bf16<<<dim3(B / 64, 1024 / 128), 256, 0, stream>>>(
        Amean, F2, WhcT, nullptr, hc_raw, 1024, F2);
    init_state<<<(B * H) / 256, 256, 0, stream>>>(hc_raw, b_h, b_c, h_state, c_state, Abuf);

    // ---- recurrent steps ----
    for (int t = 0; t < T; ++t) {
        attn_softmax<<<B, 256, 0, stream>>>(h_state, w_a_h, featsc, alfa, alfas, t);
        if (use_bf_input)
            context_kernel_bf<<<B, 256, 0, stream>>>(input_bf, alfa, Abuf);
        else
            context_kernel_f32<<<dim3(B, F2 / 256), 256, 0, stream>>>(input_data, alfa, Abuf);
        word_convert<<<(B * D) / 256, 256, 0, stream>>>(word_inputs, Abuf, t);
        // z = Abuf[128,1280] x WcT^T[1280,2048] + lstm_bias
        wmma_gemm_bf16<<<dim3(B / 64, N4H / 128), 256, 0, stream>>>(
            Abuf, KD, WcT, lstm_bias, zbuf, N4H, KD);
        gate_kernel<<<(B * H) / 256, 256, 0, stream>>>(
            zbuf, c_state, h_state, Abuf, hiddens, memory, t);
    }
}
